// MTAttention_v2_28321014349902
// MI455X (gfx1250) — matrix operand order repaired and checked by reading, not compiled, not run on hardware
//
#include <hip/hip_runtime.h>
#include <hip/hip_bf16.h>
#include <cstdint>
#include <cstddef>

// ---------------------------------------------------------------------------
// MTAttention_v2 for MI455X (gfx1250, wave32, WMMA)
// Dominant work: x(46656x512) @ [v_w;k_w]^T (512x1024)  ~49 GFLOP
// -> pre-convert to bf16 once, then bf16 WMMA (v_wmma_f32_16x16x32_bf16)
//    with f32 accumulation; LDS staging via GLOBAL_LOAD_ASYNC_TO_LDS_B128.
// Workspace requirement: ~252 MB (z 191 MB + x16 48 MB dominate).
// ---------------------------------------------------------------------------

typedef __attribute__((ext_vector_type(16))) __bf16 v16bf;
typedef __attribute__((ext_vector_type(8)))  float  v8f;

union FragBF16 {
    uint4 u[2];
    v16bf v;
};

__device__ __forceinline__ unsigned short f2bf(float f) {
    unsigned u = __float_as_uint(f);
    // round-to-nearest-even
    return (unsigned short)((u + 0x7FFFu + ((u >> 16) & 1u)) >> 16);
}
__device__ __forceinline__ unsigned pack2(float a, float b) {
    return (unsigned)f2bf(a) | ((unsigned)f2bf(b) << 16);
}

static constexpr int    BB    = 64;
static constexpr int    NN    = 729;   // 27*27
static constexpr int    DD    = 512;
static constexpr int    CC    = 512;
static constexpr int    HEADS = 8;
static constexpr size_t MROWS = (size_t)BB * NN;   // 46656
static constexpr size_t CT    = 1024;              // concatenated V|K channels

// ---------------------------------------------------------------------------
// One-shot f32 -> bf16 conversion (bandwidth bound, reads x once)
// ---------------------------------------------------------------------------
__global__ __launch_bounds__(256) void convert_x_bf16(
    const float* __restrict__ x, unsigned short* __restrict__ x16, int n8)
{
    int i = blockIdx.x * 256 + threadIdx.x;
    if (i >= n8) return;
    const float4* src = (const float4*)x + (size_t)i * 2;
    float4 f0 = src[0], f1 = src[1];
    uint4 pk = { pack2(f0.x, f0.y), pack2(f0.z, f0.w),
                 pack2(f1.x, f1.y), pack2(f1.z, f1.w) };
    ((uint4*)x16)[i] = pk;
}

__global__ __launch_bounds__(256) void convert_w_bf16(
    const float* __restrict__ vw, const float* __restrict__ kw,
    unsigned short* __restrict__ w16, int n8)
{
    int i = blockIdx.x * 256 + threadIdx.x;
    if (i >= n8) return;
    int row = i >> 6;                 // 512 f32 per row -> 64 chunks of 8
    int off = (i & 63) << 3;
    const float* src = ((row < CC) ? (vw + (size_t)row * DD)
                                   : (kw + (size_t)(row - CC) * DD)) + off;
    float4 f0 = ((const float4*)src)[0];
    float4 f1 = ((const float4*)src)[1];
    uint4 pk = { pack2(f0.x, f0.y), pack2(f0.z, f0.w),
                 pack2(f1.x, f1.y), pack2(f1.z, f1.w) };
    ((uint4*)w16)[i] = pk;
}

// ---------------------------------------------------------------------------
// GEMM: z[46656 x 1024] = x16 @ w16^T  (bf16 WMMA, f32 accum)
// Block tile 64(M) x 128(N), 8 waves as 2x4, each wave 32x32 (2x2 WMMA tiles)
// LDS staging uses GLOBAL_LOAD_ASYNC_TO_LDS_B128 (ASYNCcnt) — no VGPR copies.
// ---------------------------------------------------------------------------
__global__ __launch_bounds__(256) void gemm_bf16_wmma(
    const unsigned short* __restrict__ x16,
    const unsigned short* __restrict__ w16, float* __restrict__ z)
{
    // LDS tiles, row stride 40 bf16 (32 data + 8 pad) to spread banks
    __shared__ alignas(16) unsigned short xa[64 * 40];
    __shared__ alignas(16) unsigned short wb[128 * 40];

    const int t     = threadIdx.x;
    const int mBase = blockIdx.x * 64;
    const int nBase = blockIdx.y * 128;
    const int lane  = t & 31;
    const int w     = t >> 5;
    const int wm    = w >> 2;          // 0..1  (rows)
    const int wn    = w & 3;           // 0..3  (cols)
    const int lrow  = lane & 15;
    const int lhalf = lane >> 4;

    // per-thread async staging addresses (fixed row/col, K advances)
    const int xr = t >> 2;             // 0..63 row of x tile
    const int xq = (t & 3) << 3;       // 0,8,16,24 (bf16 elems)
    const unsigned short* xg = x16 + (size_t)(mBase + xr) * DD + xq;
    const unsigned xl = (unsigned)(size_t)&xa[xr * 40 + xq];   // LDS byte offset

    const int wc = t >> 1;             // 0..127 col of w tile
    const int wq = (t & 1) << 4;       // 0,16 (bf16 elems)
    const unsigned short* wg = w16 + (size_t)(nBase + wc) * DD + wq;
    const unsigned wl = (unsigned)(size_t)&wb[wc * 40 + wq];

    const v8f vzero = {0.f, 0.f, 0.f, 0.f, 0.f, 0.f, 0.f, 0.f};
    v8f acc[2][2];
    acc[0][0] = vzero; acc[0][1] = vzero; acc[1][0] = vzero; acc[1][1] = vzero;

    for (int kk = 0; kk < DD; kk += 32) {
        __syncthreads();   // previous iteration's ds_loads complete

        // async copy 64x32 x-tile and 128x32 w-tile straight into LDS
        const unsigned short* xgk = xg + kk;
        const unsigned short* wgk = wg + kk;
        asm volatile("global_load_async_to_lds_b128 %0, %1, off"
                     :: "v"(xl), "v"(xgk) : "memory");
        asm volatile("global_load_async_to_lds_b128 %0, %1, off"
                     :: "v"(wl), "v"(wgk) : "memory");
        // INST_OFFSET applies to both LDS and global addresses (ISA 10.x):
        asm volatile("global_load_async_to_lds_b128 %0, %1, off offset:16"
                     :: "v"(wl), "v"(wgk) : "memory");

        asm volatile("s_wait_asynccnt 0x0" ::: "memory");
        __syncthreads();

        // A fragments (16x32 bf16): lanes<16 -> K[0..7],[16..23]; lanes>=16 -> K[8..15],[24..31]
        FragBF16 a[2], b[2];
#pragma unroll
        for (int mt = 0; mt < 2; ++mt) {
            const unsigned short* ab = &xa[(wm * 32 + mt * 16 + lrow) * 40];
            a[mt].u[0] = *(const uint4*)(ab + lhalf * 8);
            a[mt].u[1] = *(const uint4*)(ab + 16 + lhalf * 8);
        }
        // B fragments (32x16 bf16): lanes<16 -> K[0..7],[16..23] of col n; lanes>=16 -> K[8..15],[24..31]
#pragma unroll
        for (int nt = 0; nt < 2; ++nt) {
            const unsigned short* bb = &wb[(wn * 32 + nt * 16 + lrow) * 40];
            b[nt].u[0] = *(const uint4*)(bb + lhalf * 8);
            b[nt].u[1] = *(const uint4*)(bb + 16 + lhalf * 8);
        }
#pragma unroll
        for (int mt = 0; mt < 2; ++mt)
#pragma unroll
            for (int nt = 0; nt < 2; ++nt)
                acc[mt][nt] = __builtin_amdgcn_wmma_f32_16x16x32_bf16(
                    false, a[mt].v, false, b[nt].v,
                    (short)0, acc[mt][nt], false, false);
    }

    // D layout: VGPR r holds (M = r + 8*lhalf, N = lrow)
#pragma unroll
    for (int mt = 0; mt < 2; ++mt)
#pragma unroll
        for (int nt = 0; nt < 2; ++nt) {
            const int row0 = mBase + wm * 32 + mt * 16 + lhalf * 8;
            const int col  = nBase + wn * 32 + nt * 16 + lrow;
#pragma unroll
            for (int r = 0; r < 8; ++r)
                z[(size_t)(row0 + r) * CT + col] = acc[mt][nt][r];
        }
}

// ---------------------------------------------------------------------------
// Per-channel sum / sum-of-squares (BN statistics), block partials + atomics
// ---------------------------------------------------------------------------
__global__ __launch_bounds__(256) void stats_kernel(const float* __restrict__ z,
                                                    float* __restrict__ sums)
{
    const int ch = blockIdx.x * 256 + threadIdx.x;       // 0..1023
    size_t r0   = (size_t)blockIdx.y * 128;
    size_t rend = r0 + 128; if (rend > MROWS) rend = MROWS;
    float s = 0.f, ss = 0.f;
    for (size_t r = r0; r < rend; ++r) {
        float v = z[r * CT + ch];
        s += v; ss += v * v;
    }
    atomicAdd(&sums[ch], s);
    atomicAdd(&sums[CT + ch], ss);
}

__global__ void finalize_kernel(const float* __restrict__ sums,
                                const float* __restrict__ vg, const float* __restrict__ vbeta,
                                const float* __restrict__ kg, const float* __restrict__ kbeta,
                                float* __restrict__ scaleC, float* __restrict__ biasC)
{
    const int ch = blockIdx.x * 256 + threadIdx.x;
    if (ch >= (int)CT) return;
    const float inv = 1.0f / (float)MROWS;
    float mu  = sums[ch] * inv;
    float var = fmaxf(sums[CT + ch] * inv - mu * mu, 0.f);
    float rs  = rsqrtf(var + 1e-5f);
    float g   = (ch < CC) ? vg[ch]    : kg[ch - CC];
    float be  = (ch < CC) ? vbeta[ch] : kbeta[ch - CC];
    scaleC[ch] = g * rs;
    biasC[ch]  = be - mu * g * rs;      // note: linear bias cancels inside BN
}

// ---------------------------------------------------------------------------
// Attention: one block per (b,h).  q = BN(k) at n=364; softmax over N=729.
// Also computes the channel-independent box averages of the attention map.
// ---------------------------------------------------------------------------
__global__ __launch_bounds__(256) void attn_kernel(
    const float* __restrict__ z, const float* __restrict__ scaleC,
    const float* __restrict__ biasC, float* __restrict__ attnW,
    float* __restrict__ avgA)
{
    __shared__ float qs[64], scs[64], bis[64];
    __shared__ float aS[NN];
    __shared__ float red[256];
    __shared__ float Sa[28 * 28];

    const int b = blockIdx.x >> 3;
    const int h = blockIdx.x & 7;
    const int t = threadIdx.x;
    const int chBase = CC + h * 64;

    if (t < 64) {
        float sc = scaleC[chBase + t], bi = biasC[chBase + t];
        scs[t] = sc; bis[t] = bi;
        qs[t]  = z[((size_t)b * NN + 364) * CT + chBase + t] * sc + bi;
    }
    __syncthreads();

    float lg[3];
    float pmax = -1e30f;
#pragma unroll
    for (int it = 0; it < 3; ++it) {
        int n = t + it * 256;
        if (n < NN) {
            const float* zr = z + ((size_t)b * NN + n) * CT + chBase;
            float d = 0.f;
            for (int dd = 0; dd < 64; ++dd)
                d += qs[dd] * (zr[dd] * scs[dd] + bis[dd]);
            d *= 0.125f;                 // dh^-0.5, dh=64
            lg[it] = d;
            pmax = fmaxf(pmax, d);
        } else lg[it] = -1e30f;
    }
    red[t] = pmax; __syncthreads();
    for (int s = 128; s > 0; s >>= 1) { if (t < s) red[t] = fmaxf(red[t], red[t + s]); __syncthreads(); }
    const float m = red[0]; __syncthreads();

    float psum = 0.f;
#pragma unroll
    for (int it = 0; it < 3; ++it) {
        int n = t + it * 256;
        if (n < NN) { float e = __expf(lg[it] - m); aS[n] = e; psum += e; }
    }
    red[t] = psum; __syncthreads();
    for (int s = 128; s > 0; s >>= 1) { if (t < s) red[t] += red[t + s]; __syncthreads(); }
    const float inv = 1.0f / red[0]; __syncthreads();

#pragma unroll
    for (int it = 0; it < 3; ++it) {
        int n = t + it * 256;
        if (n < NN) {
            float a = aS[n] * inv;
            aS[n] = a;
            attnW[(size_t)blockIdx.x * NN + n] = a;
        }
    }
    __syncthreads();

    // integral image of attn (28x28 with zero pad row/col)
    if (t < 28) { Sa[t] = 0.f; Sa[t * 28] = 0.f; }
    __syncthreads();
    if (t < 27) {
        float run = 0.f;
        for (int c = 0; c < 27; ++c) { run += aS[t * 27 + c]; Sa[(t + 1) * 28 + c + 1] = run; }
    }
    __syncthreads();
    if (t >= 1 && t < 28)
        for (int r = 2; r < 28; ++r) Sa[r * 28 + t] += Sa[(r - 1) * 28 + t];
    __syncthreads();

    if (t < 65) {
        const int i = t / 5 + 1, p = t % 5, rc = 13;
        const int r0 = (p == 2) ? rc : rc - i;
        const int r1 = (p == 1) ? rc : rc + i;
        const int c0 = (p == 4) ? rc : rc - i;
        const int c1 = (p == 3) ? rc : rc + i;
        float cnt = (float)((r1 - r0 + 1) * (c1 - c0 + 1));
        float box = Sa[(r1 + 1) * 28 + c1 + 1] - Sa[r0 * 28 + c1 + 1]
                  - Sa[(r1 + 1) * 28 + c0]     + Sa[r0 * 28 + c0];
        avgA[(size_t)blockIdx.x * 65 + t] = box / cnt;
    }
}

// ---------------------------------------------------------------------------
// Pool: one block per (b, 8 channels). Per-wave integral image + 65 boxes.
// ---------------------------------------------------------------------------
__global__ __launch_bounds__(256) void pool_kernel(
    const float* __restrict__ z, const float* __restrict__ scaleC,
    const float* __restrict__ biasC, const float* __restrict__ attnW,
    const float* __restrict__ avgA, float* __restrict__ Y)
{
    __shared__ float aS[NN];
    __shared__ float S[8][28 * 28];

    const int b = blockIdx.x, cg = blockIdx.y, t = threadIdx.x;
    const int cbase = cg * 8;
    const int h = cbase >> 6;                    // channel -> head (dh=64)
    const float* ar = attnW + (size_t)(b * HEADS + h) * NN;

    for (int n = t; n < NN; n += 256) aS[n] = ar[n];
    if (t < 224) { int ch = t / 28, c = t % 28; S[ch][c] = 0.f; S[ch][c * 28] = 0.f; }
    __syncthreads();

    // xm = attn * relu(BN(z_v)) laid into 8 padded 28x28 tiles
    for (int idx = t; idx < 8 * NN; idx += 256) {
        int ch = idx & 7, n = idx >> 3;
        int r = n / 27, cc = n % 27;
        float val = z[((size_t)b * NN + n) * CT + cbase + ch];
        float vb  = fmaxf(val * scaleC[cbase + ch] + biasC[cbase + ch], 0.f);
        S[ch][(r + 1) * 28 + cc + 1] = vb * aS[n];
    }
    __syncthreads();

    const int w = t >> 5, lane = t & 31;
    if (lane < 27) {
        float run = 0.f;
        float* row = &S[w][(lane + 1) * 28];
        for (int c = 1; c < 28; ++c) { run += row[c]; row[c] = run; }
    }
    __syncthreads();
    if (lane >= 1 && lane < 28)
        for (int r = 2; r < 28; ++r) S[w][r * 28 + lane] += S[w][(r - 1) * 28 + lane];
    __syncthreads();

    const int c = cbase + w;
    for (int j = lane; j < 65; j += 32) {
        const int i = j / 5 + 1, p = j % 5, rc = 13;
        const int r0  = (p == 2) ? rc : rc - i;
        const int r1  = (p == 1) ? rc : rc + i;
        const int cb0 = (p == 4) ? rc : rc - i;
        const int cb1 = (p == 3) ? rc : rc + i;
        float cnt = (float)((r1 - r0 + 1) * (cb1 - cb0 + 1));
        float box = S[w][(r1 + 1) * 28 + cb1 + 1] - S[w][r0 * 28 + cb1 + 1]
                  - S[w][(r1 + 1) * 28 + cb0]     + S[w][r0 * 28 + cb0];
        float ax = box / cnt;
        float aa = avgA[(size_t)(b * HEADS + h) * 65 + j];
        Y[((size_t)b * 66 + 1 + j) * CC + c] = ax / (aa + 1e-16f);
    }
    if (lane == 0) {   // center element xm[rc][rc]  (n = 13*27+13 = 364)
        float val = z[((size_t)b * NN + 364) * CT + c];
        float vb  = fmaxf(val * scaleC[c] + biasC[c], 0.f);
        Y[((size_t)b * 66) * CC + c] = aS[364] * vb;
    }
}

// ---------------------------------------------------------------------------
// Final L2 normalization over C per (b, j)
// ---------------------------------------------------------------------------
__global__ __launch_bounds__(256) void norm_kernel(const float* __restrict__ Y,
                                                   float* __restrict__ out)
{
    __shared__ float red[256];
    const size_t row = blockIdx.x;
    const int t = threadIdx.x;
    float a  = Y[row * CC + t];
    float bb = Y[row * CC + t + 256];
    red[t] = a * a + bb * bb; __syncthreads();
    for (int s = 128; s > 0; s >>= 1) { if (t < s) red[t] += red[t + s]; __syncthreads(); }
    float inv = 1.0f / fmaxf(sqrtf(red[0]), 1e-12f);
    out[row * CC + t]       = a  * inv;
    out[row * CC + t + 256] = bb * inv;
}

__global__ void zero_kernel(float* p, int n)
{
    int i = blockIdx.x * 256 + threadIdx.x;
    if (i < n) p[i] = 0.f;
}

// ---------------------------------------------------------------------------
extern "C" void kernel_launch(void* const* d_in, const int* in_sizes, int n_in,
                              void* d_out, int out_size, void* d_ws, size_t ws_size,
                              hipStream_t stream)
{
    (void)in_sizes; (void)n_in; (void)out_size; (void)ws_size;

    const float* x      = (const float*)d_in[0];
    const float* v_w    = (const float*)d_in[1];
    const float* v_g    = (const float*)d_in[3];
    const float* v_beta = (const float*)d_in[4];
    const float* k_w    = (const float*)d_in[5];
    const float* k_g    = (const float*)d_in[7];
    const float* k_beta = (const float*)d_in[8];
    float* out = (float*)d_out;

    // workspace layout (float-sized slots); ~63.1M floats (~252 MB)
    float* W = (float*)d_ws;
    const size_t Z_OFF    = 0;
    const size_t SUM_OFF  = Z_OFF + MROWS * CT;
    const size_t SC_OFF   = SUM_OFF + 2 * CT;
    const size_t BI_OFF   = SC_OFF + CT;
    const size_t ATTN_OFF = BI_OFF + CT;
    const size_t AVGA_OFF = ATTN_OFF + (size_t)BB * HEADS * NN;
    const size_t Y_OFF    = AVGA_OFF + (size_t)BB * HEADS * 65;
    const size_t X16_OFF  = Y_OFF + (size_t)BB * 66 * CC;          // bf16: 2 per float slot
    const size_t W16_OFF  = X16_OFF + (MROWS * DD) / 2;
    // end: W16_OFF + (CT*DD)/2

    float* z      = W + Z_OFF;
    float* sums   = W + SUM_OFF;
    float* scaleC = W + SC_OFF;
    float* biasC  = W + BI_OFF;
    float* attnW  = W + ATTN_OFF;
    float* avgA   = W + AVGA_OFF;
    float* Y      = W + Y_OFF;
    unsigned short* x16 = (unsigned short*)(W + X16_OFF);
    unsigned short* w16 = (unsigned short*)(W + W16_OFF);

    zero_kernel<<<(2 * (int)CT + 255) / 256, 256, 0, stream>>>(sums, 2 * (int)CT);

    // one-shot bf16 conversion (x read exactly once in f32)
    const int nx8 = (int)((MROWS * DD) / 8);
    const int nw8 = (int)((CT * DD) / 8);
    convert_x_bf16<<<(nx8 + 255) / 256, 256, 0, stream>>>(x, x16, nx8);
    convert_w_bf16<<<(nw8 + 255) / 256, 256, 0, stream>>>(v_w, k_w, w16, nw8);

    // GEMM: 729 M-tiles of 64 rows, 8 N-tiles of 128 cols (V|K concatenated)
    gemm_bf16_wmma<<<dim3(729, 8), 256, 0, stream>>>(x16, w16, z);

    stats_kernel<<<dim3(4, ((unsigned)MROWS + 127) / 128), 256, 0, stream>>>(z, sums);
    finalize_kernel<<<4, 256, 0, stream>>>(sums, v_g, v_beta, k_g, k_beta, scaleC, biasC);

    attn_kernel<<<BB * HEADS, 256, 0, stream>>>(z, scaleC, biasC, attnW, avgA);
    pool_kernel<<<dim3(BB, CC / 8), 256, 0, stream>>>(z, scaleC, biasC, attnW, avgA, Y);
    norm_kernel<<<BB * 66, 256, 0, stream>>>(Y, out);
}
